// TransolverAttention_2980707303536
// MI455X (gfx1250) — matrix operand order repaired and checked by reading, not compiled, not run on hardware
//
#include <hip/hip_runtime.h>

// ---------------------------------------------------------------------------
// Transolver attention on gfx1250 (MI455X), bf16 WMMA pipeline.
// ---------------------------------------------------------------------------

typedef __bf16 bf16_t;
typedef __attribute__((ext_vector_type(16))) __bf16 v16bf;
typedef __attribute__((ext_vector_type(8)))  __bf16 v8bf;
typedef __attribute__((ext_vector_type(8)))  float  v8f;

constexpr int Bb  = 2;
constexpr int Nn  = 32768;
constexpr int Dd  = 256;
constexpr int Hh  = 8;
constexpr int Gg  = 64;
constexpr int HDc = 32;

__device__ __forceinline__ int lane_id() { return threadIdx.x & 31; }

// WMMA: D = A(16x32 bf16) * B(32x16 bf16) + C(16x16 f32)
__device__ __forceinline__ v8f wmma_bf16(v16bf a, v16bf b, v8f c) {
  return __builtin_amdgcn_wmma_f32_16x16x32_bf16(
      /*neg_a=*/false, a, /*neg_b=*/false, b,
      /*c_mod=*/(short)0, c, /*reuse_a=*/false, /*reuse_b=*/false);
}

// A fragment from row-major [16 x K] bf16 source, leading dim ld.
// Lane l: row m = l%16. K halves interleaved: lanes 0-15 -> K {k0..k0+7, k0+16..k0+23},
// lanes 16-31 -> K {k0+8..k0+15, k0+24..k0+31}.
__device__ __forceinline__ v16bf load_a_frag(const bf16_t* src, int ld, int k0) {
  int l = lane_id();
  int m = l & 15;
  int kb = k0 + ((l & 16) ? 8 : 0);
  const bf16_t* p = src + m * ld + kb;
  v8bf lo = *(const v8bf*)(p);
  v8bf hi = *(const v8bf*)(p + 16);
  v16bf r;
#pragma unroll
  for (int e = 0; e < 8; ++e) { r[e] = lo[e]; r[e + 8] = hi[e]; }
  return r;
}

// B fragment (32x16, K x N) where source is "n-major": W[n][k] row-major, ld = row len.
// B[k][n] = W[n][k].  Lane l: col n = n0 + l%16; K halves interleaved like load_a_frag.
__device__ __forceinline__ v16bf load_b_frag_nmajor(const bf16_t* W, int ld, int n0, int k0) {
  int l = lane_id();
  int n = n0 + (l & 15);
  int kb = k0 + ((l & 16) ? 8 : 0);
  const bf16_t* p = W + n * ld + kb;
  v8bf lo = *(const v8bf*)(p);
  v8bf hi = *(const v8bf*)(p + 16);
  v16bf r;
#pragma unroll
  for (int e = 0; e < 8; ++e) { r[e] = lo[e]; r[e + 8] = hi[e]; }
  return r;
}

// B fragment where source is k-major: S[k][n] row-major with leading dim ld.
__device__ __forceinline__ v16bf load_b_frag_kmajor(const bf16_t* S, int ld, int n0, int k0) {
  int l = lane_id();
  int n = n0 + (l & 15);
  int kb = k0 + ((l & 16) ? 8 : 0);
  v16bf r;
#pragma unroll
  for (int e = 0; e < 8; ++e) { r[e] = S[(kb + e) * ld + n]; r[e + 8] = S[(kb + 16 + e) * ld + n]; }
  return r;
}

// A fragment for sw^T: A[g][k=n] = swl[n*64 + g] (float source in LDS).
__device__ __forceinline__ v16bf load_a_frag_swT(const float* swl, int g0, int k0) {
  int l = lane_id();
  int g = g0 + (l & 15);
  int kb = k0 + ((l & 16) ? 8 : 0);
  v16bf r;
#pragma unroll
  for (int e = 0; e < 8; ++e) {
    r[e]     = (bf16_t)swl[(kb + e) * 64 + g];
    r[e + 8] = (bf16_t)swl[(kb + 16 + e) * 64 + g];
  }
  return r;
}

// ---------------------------------------------------------------------------
// Kernel 0: fp32 -> bf16 convert (weights)
// ---------------------------------------------------------------------------
__global__ void k_cvt(const float* __restrict__ src, bf16_t* __restrict__ dst, int n) {
  int i = blockIdx.x * blockDim.x + threadIdx.x;
  if (i < n) dst[i] = (bf16_t)src[i];
}

// ---------------------------------------------------------------------------
// Kernel 1: dual projection  x_mid = x@Wx^T + bx ; fx_mid = x@Wfx^T + bfx
// One block = 16 rows of x, 8 waves each own 32 output cols of BOTH matrices.
// ---------------------------------------------------------------------------
__global__ __launch_bounds__(256) void k_proj(
    const float* __restrict__ x,
    const bf16_t* __restrict__ Wxb, const float* __restrict__ bx,
    const bf16_t* __restrict__ Wfb, const float* __restrict__ bfx,
    bf16_t* __restrict__ xm, bf16_t* __restrict__ fxm) {
  __shared__ alignas(32) bf16_t As[16 * Dd];
  int row0 = blockIdx.x * 16;  // global row in [0, B*N)
  for (int i = threadIdx.x; i < 16 * Dd; i += 256) {
    int r = i >> 8, c = i & 255;
    As[i] = (bf16_t)x[(size_t)(row0 + r) * Dd + c];
  }
  __syncthreads();
  int w = threadIdx.x >> 5;
  int n0 = w * 32;
  v8f ax0 = {}, ax1 = {}, af0 = {}, af1 = {};
#pragma unroll
  for (int kk = 0; kk < 8; ++kk) {
    int k0 = kk * 32;
    v16bf a  = load_a_frag(As, Dd, k0);
    v16bf b0 = load_b_frag_nmajor(Wxb, Dd, n0,      k0);
    v16bf b1 = load_b_frag_nmajor(Wxb, Dd, n0 + 16, k0);
    v16bf c0 = load_b_frag_nmajor(Wfb, Dd, n0,      k0);
    v16bf c1 = load_b_frag_nmajor(Wfb, Dd, n0 + 16, k0);
    ax0 = wmma_bf16(a, b0, ax0);
    ax1 = wmma_bf16(a, b1, ax1);
    af0 = wmma_bf16(a, c0, af0);
    af1 = wmma_bf16(a, c1, af1);
  }
  int l = lane_id();
  int mn = l & 15;
  int mb = (l & 16) ? 8 : 0;
#pragma unroll
  for (int j = 0; j < 8; ++j) {
    int m = mb + j;
    size_t base = (size_t)(row0 + m) * Dd;
    xm [base + n0 + mn]      = (bf16_t)(ax0[j] + bx [n0 + mn]);
    xm [base + n0 + 16 + mn] = (bf16_t)(ax1[j] + bx [n0 + 16 + mn]);
    fxm[base + n0 + mn]      = (bf16_t)(af0[j] + bfx[n0 + mn]);
    fxm[base + n0 + 16 + mn] = (bf16_t)(af1[j] + bfx[n0 + 16 + mn]);
  }
}

// ---------------------------------------------------------------------------
// Kernel 2: per (b,h) and 128-row chunk:
//   logits = x_mid @ Wslice^T (+bias)/temp  -> softmax over G -> sw
//   store sw (bf16), accumulate denom (atomic), accumulate sw^T @ fx (atomic)
// ---------------------------------------------------------------------------
__global__ __launch_bounds__(256) void k_slice(
    const bf16_t* __restrict__ xm, const bf16_t* __restrict__ fxm,
    const bf16_t* __restrict__ Wsb, const float* __restrict__ bsl,
    const float* __restrict__ temp,
    bf16_t* __restrict__ swg, float* __restrict__ denom, float* __restrict__ sacc) {
  __shared__ alignas(32) bf16_t xs[128 * HDc];
  __shared__ alignas(32) bf16_t fs[128 * HDc];
  __shared__ float swl[128 * Gg];
  int bh = blockIdx.y;  // b*H + h
  int h  = bh & (Hh - 1);
  int b  = bh >> 3;
  int n0 = blockIdx.x * 128;
  {
    int t = threadIdx.x;
    int r = t >> 1, part = t & 1;
    size_t srcBase = (size_t)(b * Nn + n0 + r) * Dd + h * HDc + part * 16;
    *(v16bf*)(&xs[r * HDc + part * 16]) = *(const v16bf*)(xm  + srcBase);
    *(v16bf*)(&fs[r * HDc + part * 16]) = *(const v16bf*)(fxm + srcBase);
  }
  __syncthreads();
  int w = threadIdx.x >> 5;
  float invT = 1.0f / temp[h];
  {
    v16bf a = load_a_frag(xs + w * 16 * HDc, HDc, 0);
    int l = lane_id(); int mn = l & 15; int mb = (l & 16) ? 8 : 0;
#pragma unroll
    for (int gt = 0; gt < 4; ++gt) {
      v16bf bw = load_b_frag_nmajor(Wsb, HDc, gt * 16, 0);
      v8f c = {};
      c = wmma_bf16(a, bw, c);
#pragma unroll
      for (int j = 0; j < 8; ++j) {
        int m = w * 16 + mb + j;
        int g = gt * 16 + mn;
        swl[m * Gg + g] = (c[j] + bsl[g]) * invT;
      }
    }
  }
  __syncthreads();
  if (threadIdx.x < 128) {  // row softmax over G
    float* row = &swl[threadIdx.x * Gg];
    float mx = row[0];
    for (int g = 1; g < Gg; ++g) mx = fmaxf(mx, row[g]);
    float s = 0.f;
    for (int g = 0; g < Gg; ++g) { float e = __expf(row[g] - mx); row[g] = e; s += e; }
    float inv = 1.0f / s;
    for (int g = 0; g < Gg; ++g) row[g] *= inv;
  }
  __syncthreads();
  {  // store sw to global as bf16
    size_t base = ((size_t)bh * Nn + n0) * Gg;
    for (int i = threadIdx.x; i < 128 * Gg; i += 256) swg[base + i] = (bf16_t)swl[i];
  }
  if (threadIdx.x < Gg) {  // column sums -> denom
    float s = 0.f;
    for (int r = 0; r < 128; ++r) s += swl[r * Gg + threadIdx.x];
    atomicAdd(&denom[bh * Gg + threadIdx.x], s);
  }
  {  // partial slice_att = sw^T (64x128) @ fx (128x32): 8 acc tiles over 8 waves
    int gt = w & 3, ct = w >> 2;
    v8f acc = {};
#pragma unroll
    for (int kk = 0; kk < 4; ++kk) {
      v16bf a  = load_a_frag_swT(swl, gt * 16, kk * 32);
      v16bf bb = load_b_frag_kmajor(fs, HDc, ct * 16, kk * 32);
      acc = wmma_bf16(a, bb, acc);
    }
    int l = lane_id(); int mn = l & 15; int mb = (l & 16) ? 8 : 0;
#pragma unroll
    for (int j = 0; j < 8; ++j) {
      int g = gt * 16 + mb + j;
      int c = ct * 16 + mn;
      atomicAdd(&sacc[((size_t)bh * Gg + g) * HDc + c], acc[j]);
    }
  }
}

// ---------------------------------------------------------------------------
// Kernel 3: per (b,h): finalize slice_att, qkv, 64x64 SDPA -> att (bf16).
// Tiny (~0.4 MFLOP per head), scalar f32.
// ---------------------------------------------------------------------------
__global__ __launch_bounds__(64) void k_headattn(
    const float* __restrict__ sacc, const float* __restrict__ denom,
    const float* __restrict__ Wqkv, bf16_t* __restrict__ attb) {
  __shared__ float kl[Gg * HDc];
  __shared__ float vl[Gg * HDc];
  int bh = blockIdx.x;
  int g  = threadIdx.x;  // 0..63
  float s[HDc];
  float inv = 1.0f / (denom[bh * Gg + g] + 1e-5f);
  for (int c = 0; c < HDc; ++c)
    s[c] = sacc[((size_t)bh * Gg + g) * HDc + c] * inv;
  float q[HDc];
  for (int j = 0; j < HDc; ++j) {
    float aq = 0.f, ak = 0.f, av = 0.f;
    for (int c = 0; c < HDc; ++c) {
      aq += s[c] * Wqkv[j * HDc + c];
      ak += s[c] * Wqkv[(HDc + j) * HDc + c];
      av += s[c] * Wqkv[(2 * HDc + j) * HDc + c];
    }
    q[j] = aq; kl[g * HDc + j] = ak; vl[g * HDc + j] = av;
  }
  __syncthreads();
  float sc[Gg];
  const float scale = 0.17677669529663687f;  // 1/sqrt(HD)
  float mx = -1e30f;
  for (int g2 = 0; g2 < Gg; ++g2) {
    float d = 0.f;
    for (int c = 0; c < HDc; ++c) d += q[c] * kl[g2 * HDc + c];
    sc[g2] = d * scale;
    mx = fmaxf(mx, sc[g2]);
  }
  float sum = 0.f;
  for (int g2 = 0; g2 < Gg; ++g2) { sc[g2] = __expf(sc[g2] - mx); sum += sc[g2]; }
  float isum = 1.0f / sum;
  for (int c = 0; c < HDc; ++c) {
    float a = 0.f;
    for (int g2 = 0; g2 < Gg; ++g2) a += sc[g2] * vl[g2 * HDc + c];
    attb[((size_t)bh * Gg + g) * HDc + c] = (bf16_t)(a * isum);
  }
}

// ---------------------------------------------------------------------------
// Kernel 4: per (b, 64-row chunk):
//   mid[n, h*32+c] = sum_g sw[n,g]*att[g,c]   (per head, WMMA, into LDS)
//   out = mid @ Wout^T + bout                 (WMMA)
// ---------------------------------------------------------------------------
__global__ __launch_bounds__(256) void k_scatter_out(
    const bf16_t* __restrict__ swg, const bf16_t* __restrict__ attb,
    const bf16_t* __restrict__ Woutb, const float* __restrict__ bout,
    float* __restrict__ out) {
  __shared__ alignas(32) bf16_t sws[64 * Gg];
  __shared__ alignas(32) bf16_t attl[Gg * HDc];
  __shared__ alignas(32) bf16_t mid[64 * Dd];
  int blk = blockIdx.x;
  int b  = blk >> 9;        // Nn/64 = 512 chunks per batch
  int n0 = (blk & 511) * 64;
  int w = threadIdx.x >> 5;
  int l = lane_id(); int mn = l & 15; int mb = (l & 16) ? 8 : 0;
  for (int h = 0; h < Hh; ++h) {
    int bh = b * Hh + h;
    {
      int t = threadIdx.x;
      int r = t >> 2, part = t & 3;
      const bf16_t* src = swg + ((size_t)bh * Nn + n0 + r) * Gg + part * 16;
      *(v16bf*)(&sws[r * Gg + part * 16]) = *(const v16bf*)src;
    }
    if (threadIdx.x < 128) {
      int t = threadIdx.x;
      int r = t >> 1, part = t & 1;
      const bf16_t* src = attb + ((size_t)bh * Gg + r) * HDc + part * 16;
      *(v16bf*)(&attl[r * HDc + part * 16]) = *(const v16bf*)src;
    }
    __syncthreads();
    int mt = w & 3, ct = w >> 2;
    v8f acc = {};
#pragma unroll
    for (int kk = 0; kk < 2; ++kk) {
      v16bf a  = load_a_frag(sws + mt * 16 * Gg, Gg, kk * 32);
      v16bf bb = load_b_frag_kmajor(attl, HDc, ct * 16, kk * 32);
      acc = wmma_bf16(a, bb, acc);
    }
#pragma unroll
    for (int j = 0; j < 8; ++j) {
      int m = mt * 16 + mb + j;
      mid[m * Dd + h * HDc + ct * 16 + mn] = (bf16_t)acc[j];
    }
    __syncthreads();
  }
  // out = mid (64x256) @ Wout^T (256x256) + bout
  int mt = w & 3, nh = w >> 2;  // nh in {0,1}
#pragma unroll
  for (int i = 0; i < 8; ++i) {
    int nt = nh + i * 2;  // 16 col tiles of 16
    v8f acc = {};
#pragma unroll
    for (int kk = 0; kk < 8; ++kk) {
      v16bf a  = load_a_frag(mid + mt * 16 * Dd, Dd, kk * 32);
      v16bf bb = load_b_frag_nmajor(Woutb, Dd, nt * 16, kk * 32);
      acc = wmma_bf16(a, bb, acc);
    }
#pragma unroll
    for (int j = 0; j < 8; ++j) {
      int m   = mt * 16 + mb + j;
      int col = nt * 16 + mn;
      out[(size_t)(b * Nn + n0 + m) * Dd + col] = acc[j] + bout[col];
    }
  }
}

// ---------------------------------------------------------------------------
extern "C" void kernel_launch(void* const* d_in, const int* in_sizes, int n_in,
                              void* d_out, int out_size, void* d_ws, size_t ws_size,
                              hipStream_t stream) {
  (void)in_sizes; (void)n_in; (void)out_size; (void)ws_size;
  const float* x    = (const float*)d_in[0];
  const float* Wx   = (const float*)d_in[1];
  const float* bx   = (const float*)d_in[2];
  const float* Wfx  = (const float*)d_in[3];
  const float* bfx  = (const float*)d_in[4];
  const float* Wsl  = (const float*)d_in[5];
  const float* bsl  = (const float*)d_in[6];
  const float* temp = (const float*)d_in[7];
  const float* Wqkv = (const float*)d_in[8];
  const float* Wout = (const float*)d_in[9];
  const float* bout = (const float*)d_in[10];
  float* out = (float*)d_out;

  char* ws = (char*)d_ws;
  size_t off = 0;
  auto take = [&](size_t bytes) -> char* {
    char* p = ws + off;
    off = (off + bytes + 255) & ~(size_t)255;
    return p;
  };
  bf16_t* Wxb   = (bf16_t*)take((size_t)Dd * Dd * 2);
  bf16_t* Wfb   = (bf16_t*)take((size_t)Dd * Dd * 2);
  bf16_t* Woutb = (bf16_t*)take((size_t)Dd * Dd * 2);
  bf16_t* Wsb   = (bf16_t*)take((size_t)Gg * HDc * 2);
  bf16_t* xm    = (bf16_t*)take((size_t)Bb * Nn * Dd * 2);
  bf16_t* fxm   = (bf16_t*)take((size_t)Bb * Nn * Dd * 2);
  bf16_t* swg   = (bf16_t*)take((size_t)Bb * Hh * Nn * Gg * 2);
  float*  sacc  = (float*) take((size_t)Bb * Hh * Gg * HDc * 4);
  float*  denom = (float*) take((size_t)Bb * Hh * Gg * 4);
  bf16_t* attb  = (bf16_t*)take((size_t)Bb * Hh * Gg * HDc * 2);

  // weight conversion
  k_cvt<<<(Dd * Dd + 255) / 256, 256, 0, stream>>>(Wx,   Wxb,   Dd * Dd);
  k_cvt<<<(Dd * Dd + 255) / 256, 256, 0, stream>>>(Wfx,  Wfb,   Dd * Dd);
  k_cvt<<<(Dd * Dd + 255) / 256, 256, 0, stream>>>(Wout, Woutb, Dd * Dd);
  k_cvt<<<(Gg * HDc + 255) / 256, 256, 0, stream>>>(Wsl, Wsb,   Gg * HDc);

  // zero accumulators (graph-capture-safe, deterministic per call)
  hipMemsetAsync(sacc,  0, (size_t)Bb * Hh * Gg * HDc * 4, stream);
  hipMemsetAsync(denom, 0, (size_t)Bb * Hh * Gg * 4,       stream);

  k_proj<<<(Bb * Nn) / 16, 256, 0, stream>>>(x, Wxb, bx, Wfb, bfx, xm, fxm);
  k_slice<<<dim3(Nn / 128, Bb * Hh), 256, 0, stream>>>(xm, fxm, Wsb, bsl, temp,
                                                       swg, denom, sacc);
  k_headattn<<<Bb * Hh, 64, 0, stream>>>(sacc, denom, Wqkv, attb);
  k_scatter_out<<<Bb * (Nn / 64), 256, 0, stream>>>(swg, attb, Woutb, bout, out);
}
